// AnomalyTransformer_68186900791524
// MI455X (gfx1250) — matrix operand order repaired and checked by reading, not compiled, not run on hardware
//
#include <hip/hip_runtime.h>
#include <hip/hip_bf16.h>
#include <math.h>

#define B_  64
#define T_  8000
#define C_  55
#define CL_ 80
#define W_  100
#define L_  101
#define D_  512
#define H_  8
#define E_  64

typedef __attribute__((ext_vector_type(16))) __bf16 v16bf;
typedef __attribute__((ext_vector_type(8)))  float  v8f;

__device__ __forceinline__ float gelu_exact(float x) {
    return 0.5f * x * (1.f + erff(x * 0.70710678118654752440f));
}

// ---------------------------------------------------------------------------
// Generic batched GEMM: C = A[M,K] * op(B) + bias, op(B)=B[K,N] (TB=0) or
// B^T with B stored [N,K] (TB=1). bf16 WMMA 16x16x32, f32 accumulate.
// One wave owns a 32x64 C tile: 2 A fragments x 4 B fragments -> 8 WMMAs per
// 32-wide k-step. All fragments are loaded (branch-free, coalescable b128)
// before the WMMA burst so the bf16 WMMA->VALU WAR hazard slots are covered
// by the next iteration's loads instead of v_nops.
// Fragment layouts used by this code (wave32; A/B share one contiguous k order):
//   A (16x32 bf16): lane = m + 16*hi ; elem j -> k = 16*hi + j
//       => afrag[0..15] = A[row, kb+16*hi .. +15]
//   B (32x16 bf16): lane = n + 16*hi ; elem j -> k = 16*hi + j
//       => TB=1: 16 contiguous floats from B[n, kb+16*hi ..]
//   C (16x16 f32):  lane = n + 16*hi ; elem j -> M = j + 8*hi
// Main K loop is guard-free; the ragged tail (only K=101 attnV) uses clamped
// addresses + select-zero, no branches.
// ---------------------------------------------------------------------------
template <int TB>
__global__ void __launch_bounds__(256)
gemm_bf16_wmma(const float* __restrict__ A, int lda, long long sAy, long long sAz,
               const float* __restrict__ Bm, int ldb, long long sBy, long long sBz,
               float* __restrict__ Cm, int ldc, long long sCy, long long sCz,
               int M, int N, int K,
               const float* __restrict__ bias, int act)
{
    const float* Ab = A  + (long long)blockIdx.y * sAy + (long long)blockIdx.z * sAz;
    const float* Bb = Bm + (long long)blockIdx.y * sBy + (long long)blockIdx.z * sBz;
    float*       Cb = Cm + (long long)blockIdx.y * sCy + (long long)blockIdx.z * sCz;

    int wave = threadIdx.x >> 5;
    int lane = threadIdx.x & 31;
    int tilesN = (N + 63) >> 6;
    int tilesM = (M + 31) >> 5;
    int tile = blockIdx.x * 8 + wave;
    if (tile >= tilesM * tilesN) return;   // uniform per wave -> EXEC all-1 for WMMA
    int mt = tile / tilesN;
    int nt = tile - mt * tilesN;
    int m0 = mt << 5, n0 = nt << 6;
    int hi = lane >> 4, ln = lane & 15;

    const float* ArowP[2];
#pragma unroll
    for (int i = 0; i < 2; ++i) {
        int mrow = m0 + (i << 4) + ln;
        int arow = mrow < M ? mrow : M - 1;   // clamp; garbage rows never stored
        ArowP[i] = Ab + (long long)arow * lda;
    }

    int colc[4], bcolc[4];
#pragma unroll
    for (int s = 0; s < 4; ++s) {
        colc[s]  = n0 + (s << 4) + ln;
        bcolc[s] = colc[s] < N ? colc[s] : N - 1;
    }

    v8f acc[2][4];
#pragma unroll
    for (int i = 0; i < 2; ++i)
#pragma unroll
        for (int s = 0; s < 4; ++s)
#pragma unroll
            for (int j = 0; j < 8; ++j) acc[i][s][j] = 0.f;

    const int kfull = K & ~31;

    for (int kb = 0; kb < kfull; kb += 32) {
        v16bf afr[2];
#pragma unroll
        for (int i = 0; i < 2; ++i) {
            const float* ap = ArowP[i] + kb + (hi << 4);
#pragma unroll
            for (int j = 0; j < 8; ++j) {
                afr[i][j]     = (__bf16)ap[j];
                afr[i][j + 8] = (__bf16)ap[8 + j];
            }
        }
        v16bf bfr[4];
#pragma unroll
        for (int s = 0; s < 4; ++s) {
            if (TB) {
                const float* bp = Bb + (long long)bcolc[s] * ldb + kb + (hi << 4);
#pragma unroll
                for (int j = 0; j < 16; ++j) bfr[s][j] = (__bf16)bp[j];
            } else {
                const float* bp = Bb + (long long)(kb + (hi << 4)) * ldb + bcolc[s];
#pragma unroll
                for (int j = 0; j < 16; ++j) bfr[s][j] = (__bf16)bp[(long long)j * ldb];
            }
        }
#pragma unroll
        for (int i = 0; i < 2; ++i)
#pragma unroll
            for (int s = 0; s < 4; ++s)
                acc[i][s] = __builtin_amdgcn_wmma_f32_16x16x32_bf16(
                    false, afr[i], false, bfr[s], (short)0, acc[i][s], false, false);
    }

    if (kfull < K) {       // ragged tail: clamped addresses + select-zero
        const int kb = kfull;
        v16bf afr[2];
#pragma unroll
        for (int i = 0; i < 2; ++i)
#pragma unroll
            for (int j = 0; j < 16; ++j) {
                int kk = kb + (hi << 4) + j;
                int kc = kk < K ? kk : K - 1;
                float av = ArowP[i][kc];
                afr[i][j] = (__bf16)(kk < K ? av : 0.f);
            }
        v16bf bfr[4];
#pragma unroll
        for (int s = 0; s < 4; ++s)
#pragma unroll
            for (int j = 0; j < 16; ++j) {
                int kk = kb + (hi << 4) + j;
                int kc = kk < K ? kk : K - 1;
                float bv = TB ? Bb[(long long)bcolc[s] * ldb + kc]
                              : Bb[(long long)kc * ldb + bcolc[s]];
                bfr[s][j] = (__bf16)(kk < K ? bv : 0.f);
            }
#pragma unroll
        for (int i = 0; i < 2; ++i)
#pragma unroll
            for (int s = 0; s < 4; ++s)
                acc[i][s] = __builtin_amdgcn_wmma_f32_16x16x32_bf16(
                    false, afr[i], false, bfr[s], (short)0, acc[i][s], false, false);
    }

#pragma unroll
    for (int s = 0; s < 4; ++s) {
        int col = colc[s];
        if (col >= N) continue;
        float bv = bias ? bias[col] : 0.f;
#pragma unroll
        for (int i = 0; i < 2; ++i)
#pragma unroll
            for (int j = 0; j < 8; ++j) {
                int row = m0 + (i << 4) + j + (hi << 3);
                if (row < M) {
                    float v = acc[i][s][j] + bv;
                    if (act == 1)      v = gelu_exact(v);
                    else if (act == 2) v = fmaxf(v, 0.f);
                    Cb[(long long)row * ldc + col] = v;
                }
            }
    }
}

// ---------------------------------------------------------------------------
// Deformable embedding: one block per (w, b) token.
//   offsets = conv(x tile) ; bilinear sample along T ; project to D=512.
// Writes raw emb into e rows 1..100 (cls/pos added by posemb_kernel).
// ---------------------------------------------------------------------------
__global__ void __launch_bounds__(256)
deform_embed_kernel(const float* __restrict__ x,
                    const float* __restrict__ offw, const float* __restrict__ offb,
                    const float* __restrict__ dw,   const float* __restrict__ db,
                    float* __restrict__ e)
{
    int w = blockIdx.x, b = blockIdx.y, tid = threadIdx.x;
    __shared__ float tile[CL_ * C_];   // [k][c]
    __shared__ float samp[C_ * CL_];   // [c][k]
    __shared__ float offv[2 * CL_];
    __shared__ float w0a[CL_], w1a[CL_];
    __shared__ int   i0a[CL_], i1a[CL_];

    const float* xb = x + (long long)b * T_ * C_;
    for (int i = tid; i < CL_ * C_; i += 256)
        tile[i] = xb[(long long)(w * CL_ + i / C_) * C_ + (i % C_)];
    __syncthreads();

    for (int o = tid; o < 2 * CL_; o += 256) {
        float acc = offb[o];
        const float* ow = offw + (long long)o * C_ * CL_;
        for (int c = 0; c < C_; ++c)
            for (int kk = 0; kk < CL_; ++kk)
                acc += tile[kk * C_ + c] * ow[c * CL_ + kk];
        offv[o] = acc;
    }
    __syncthreads();

    if (tid < CL_) {
        float dy = offv[2 * tid], dx = offv[2 * tid + 1];
        float xs = (float)(w * CL_ + tid) + dx;
        float wy = fmaxf(0.f, 1.f - fabsf(dy));
        float xf = floorf(xs);
        float lx = xs - xf;
        int ix0 = (int)xf;
        float m0 = (ix0 >= 0     && ix0 < T_)     ? 1.f : 0.f;
        float m1 = (ix0 + 1 >= 0 && ix0 + 1 < T_) ? 1.f : 0.f;
        w0a[tid] = wy * (1.f - lx) * m0;
        w1a[tid] = wy * lx * m1;
        i0a[tid] = min(max(ix0, 0), T_ - 1);
        i1a[tid] = min(max(ix0 + 1, 0), T_ - 1);
    }
    __syncthreads();

    for (int i = tid; i < C_ * CL_; i += 256) {
        int c = i / CL_, kk = i % CL_;
        samp[i] = w0a[kk] * xb[(long long)i0a[kk] * C_ + c]
                + w1a[kk] * xb[(long long)i1a[kk] * C_ + c];
    }
    __syncthreads();

    for (int d = tid; d < D_; d += 256) {
        float acc = db[d];
        const float* dwr = dw + (long long)d * C_ * CL_;
        for (int i = 0; i < C_ * CL_; ++i) acc += samp[i] * dwr[i];
        e[((long long)b * L_ + 1 + w) * D_ + d] = acc;
    }
}

// e[b,l,:] += pe[l,:]; row 0 initialized from cls token.
__global__ void posemb_kernel(float* __restrict__ e, const float* __restrict__ cls)
{
    long long idx = (long long)blockIdx.x * 256 + threadIdx.x;
    if (idx >= (long long)B_ * L_ * D_) return;
    int d = (int)(idx & (D_ - 1));
    long long row = idx >> 9;
    int l = (int)(row % L_);
    float div = expf(-(float)(d & ~1) * (9.210340371976184f / 512.f)); // ln(10000)/D
    float arg = (float)l * div;
    float pe = (d & 1) ? cosf(arg) : sinf(arg);
    float base = (l == 0) ? cls[d] : e[idx];
    e[idx] = base + pe;
}

// Row softmax over L=101 with pre-scale 1/8, in place.
__global__ void __launch_bounds__(128)
softmax_kernel(float* __restrict__ p)
{
    float* r = p + (long long)blockIdx.x * L_;
    int t = threadIdx.x;
    __shared__ float red[128];
    float v = (t < L_) ? r[t] * 0.125f : -3.0e38f;
    red[t] = v; __syncthreads();
    for (int s = 64; s > 0; s >>= 1) { if (t < s) red[t] = fmaxf(red[t], red[t + s]); __syncthreads(); }
    float mx = red[0]; __syncthreads();
    float ev = (t < L_) ? expf(v - mx) : 0.f;
    red[t] = ev; __syncthreads();
    for (int s = 64; s > 0; s >>= 1) { if (t < s) red[t] += red[t + s]; __syncthreads(); }
    float inv = 1.f / red[0];
    if (t < L_) r[t] = ev * inv;
}

// sigma transform + Gaussian prior row; one block per (b,h,l).
__global__ void __launch_bounds__(128)
prior_sigma_kernel(const float* __restrict__ siglin,
                   float* __restrict__ prior, float* __restrict__ sigout)
{
    int id = blockIdx.x;
    int l = id % L_;
    int h = (id / L_) % H_;
    int b = id / (L_ * H_);
    int t = threadIdx.x;
    float sraw = siglin[((long long)b * L_ + l) * H_ + h];
    float sg  = 1.f / (1.f + expf(-5.f * sraw)) + 1e-5f;
    float sig = expf(sg * 1.0986122886681098f) - 1.f;   // 3^sg - 1
    float coef   = 0.3989422804014327f / sig;           // 1/(sqrt(2pi)*sig)
    float inv2s2 = 0.5f / (sig * sig);
    float* pr = prior  + (long long)id * L_;
    float* so = sigout + (long long)id * L_;
    if (t < L_) {
        float dd = (float)(l - t);
        pr[t] = coef * expf(-dd * dd * inv2s2);
        so[t] = sig;
    }
}

// out = LayerNorm(A + Bv) * s + bb, row length 512, 256 threads x 2 elems.
__global__ void __launch_bounds__(256)
resid_ln_kernel(const float* __restrict__ A, const float* __restrict__ Bv,
                const float* __restrict__ s, const float* __restrict__ bb,
                float* __restrict__ out)
{
    int r = blockIdx.x, t = threadIdx.x;
    const float* a = A  + (long long)r * D_;
    const float* g = Bv + (long long)r * D_;
    float x0 = a[t] + g[t], x1 = a[t + 256] + g[t + 256];
    __shared__ float red[256];
    red[t] = x0 + x1; __syncthreads();
    for (int st = 128; st > 0; st >>= 1) { if (t < st) red[t] += red[t + st]; __syncthreads(); }
    float mean = red[0] * (1.f / 512.f); __syncthreads();
    float d0 = x0 - mean, d1 = x1 - mean;
    red[t] = d0 * d0 + d1 * d1; __syncthreads();
    for (int st = 128; st > 0; st >>= 1) { if (t < st) red[t] += red[t + st]; __syncthreads(); }
    float rstd = rsqrtf(red[0] * (1.f / 512.f) + 1e-5f);
    out[(long long)r * D_ + t]       = d0 * rstd * s[t]       + bb[t];
    out[(long long)r * D_ + t + 256] = d1 * rstd * s[t + 256] + bb[t + 256];
}

// final LN -> head LN -> relu(FC 512->256) -> FC 256->2 ; one block per batch b.
__global__ void __launch_bounds__(256)
head_kernel(const float* __restrict__ e,
            const float* __restrict__ fs, const float* __restrict__ fb,
            const float* __restrict__ hs, const float* __restrict__ hb,
            const float* __restrict__ w1, const float* __restrict__ b1,
            const float* __restrict__ w2, const float* __restrict__ b2,
            float* __restrict__ out)
{
    int b = blockIdx.x, t = threadIdx.x;
    __shared__ float row[D_];
    __shared__ float red[256];
    __shared__ float h1[256];
    const float* er = e + (long long)b * L_ * D_;   // l = 0 row
    float x0 = er[t], x1 = er[t + 256];
    for (int pass = 0; pass < 2; ++pass) {
        red[t] = x0 + x1; __syncthreads();
        for (int s = 128; s > 0; s >>= 1) { if (t < s) red[t] += red[t + s]; __syncthreads(); }
        float mean = red[0] * (1.f / 512.f); __syncthreads();
        float d0 = x0 - mean, d1 = x1 - mean;
        red[t] = d0 * d0 + d1 * d1; __syncthreads();
        for (int s = 128; s > 0; s >>= 1) { if (t < s) red[t] += red[t + s]; __syncthreads(); }
        float rstd = rsqrtf(red[0] * (1.f / 512.f) + 1e-5f); __syncthreads();
        const float* ss = pass == 0 ? fs : hs;
        const float* sb = pass == 0 ? fb : hb;
        x0 = d0 * rstd * ss[t]       + sb[t];
        x1 = d1 * rstd * ss[t + 256] + sb[t + 256];
    }
    row[t] = x0; row[t + 256] = x1;
    __syncthreads();
    float acc = b1[t];
    const float* wr = w1 + (long long)t * D_;
    for (int i = 0; i < D_; ++i) acc += row[i] * wr[i];
    h1[t] = fmaxf(acc, 0.f);
    __syncthreads();
    if (t < 2) {
        float a2 = b2[t];
        const float* w2r = w2 + t * 256;
        for (int i = 0; i < 256; ++i) a2 += h1[i] * w2r[i];
        out[b * 2 + t] = a2;
    }
}

// ---------------------------------------------------------------------------
// Host orchestration. Input order assumes recursive insertion-order flatten of
// setup_inputs(): x, offset_w, offset_b, deform_w, deform_b, cls_token,
// 2 x [wq,bq,wk,bk,wv,bv,wsig,bsig,wo,bo,w1,b1,w2,b2,ln1_s,ln1_b,ln2_s,ln2_b],
// final_ln_s, final_ln_b, head_ln_s, head_ln_b, head_w1, head_b1, head_w2, head_b2.
// ---------------------------------------------------------------------------
static inline int gemm_blocks(int M, int N) {
    int tiles = ((M + 31) / 32) * ((N + 63) / 64);
    return (tiles + 7) / 8;
}

extern "C" void kernel_launch(void* const* d_in, const int* in_sizes, int n_in,
                              void* d_out, int out_size, void* d_ws, size_t ws_size,
                              hipStream_t stream)
{
    (void)in_sizes; (void)n_in; (void)out_size; (void)ws_size;

    const float* x    = (const float*)d_in[0];
    const float* offw = (const float*)d_in[1];
    const float* offb = (const float*)d_in[2];
    const float* dw   = (const float*)d_in[3];
    const float* db   = (const float*)d_in[4];
    const float* cls  = (const float*)d_in[5];
    const float* fin_s = (const float*)d_in[42];
    const float* fin_b = (const float*)d_in[43];
    const float* hln_s = (const float*)d_in[44];
    const float* hln_b = (const float*)d_in[45];
    const float* hw1   = (const float*)d_in[46];
    const float* hb1   = (const float*)d_in[47];
    const float* hw2   = (const float*)d_in[48];
    const float* hb2   = (const float*)d_in[49];

    float* out = (float*)d_out;
    float* ws  = (float*)d_ws;
    const long long BLD = (long long)B_ * L_ * D_;   // 3,309,568
    float* buf0 = ws;               // e
    float* buf1 = ws + 1 * BLD;     // q / attnV / ffn-hidden
    float* buf2 = ws + 2 * BLD;     // k / new_x / ffn-out
    float* buf3 = ws + 3 * BLD;     // v / xr
    float* siglin = ws + 4 * BLD;   // [B,L,H]

    const long long PLL = (long long)B_ * H_ * L_ * L_;  // per-layer [B,H,L,L]
    float* series_base = out + 128;
    float* prior_base  = series_base + 2 * PLL;
    float* sigma_base  = prior_base  + 2 * PLL;

    deform_embed_kernel<<<dim3(W_, B_), 256, 0, stream>>>(x, offw, offb, dw, db, buf0);
    posemb_kernel<<<(unsigned)((BLD + 255) / 256), 256, 0, stream>>>(buf0, cls);

    const int MBL = B_ * L_;   // 6464 = 202 * 32
    for (int layer = 0; layer < 2; ++layer) {
        int p = 6 + layer * 18;
        const float* wq   = (const float*)d_in[p + 0];
        const float* bq   = (const float*)d_in[p + 1];
        const float* wk   = (const float*)d_in[p + 2];
        const float* bk   = (const float*)d_in[p + 3];
        const float* wv   = (const float*)d_in[p + 4];
        const float* bv   = (const float*)d_in[p + 5];
        const float* wsig = (const float*)d_in[p + 6];
        const float* bsig = (const float*)d_in[p + 7];
        const float* wo   = (const float*)d_in[p + 8];
        const float* bo   = (const float*)d_in[p + 9];
        const float* w1   = (const float*)d_in[p + 10];
        const float* b1   = (const float*)d_in[p + 11];
        const float* w2   = (const float*)d_in[p + 12];
        const float* b2   = (const float*)d_in[p + 13];
        const float* l1s  = (const float*)d_in[p + 14];
        const float* l1b  = (const float*)d_in[p + 15];
        const float* l2s  = (const float*)d_in[p + 16];
        const float* l2b  = (const float*)d_in[p + 17];

        float* series_l = series_base + (long long)layer * PLL;
        float* prior_l  = prior_base  + (long long)layer * PLL;
        float* sigma_l  = sigma_base  + (long long)layer * PLL;

        // q, k, v : [6464,512] = e @ W^T + b
        gemm_bf16_wmma<1><<<dim3(gemm_blocks(MBL, D_)), 256, 0, stream>>>(
            buf0, D_, 0, 0, wq, D_, 0, 0, buf1, D_, 0, 0, MBL, D_, D_, bq, 0);
        gemm_bf16_wmma<1><<<dim3(gemm_blocks(MBL, D_)), 256, 0, stream>>>(
            buf0, D_, 0, 0, wk, D_, 0, 0, buf2, D_, 0, 0, MBL, D_, D_, bk, 0);
        gemm_bf16_wmma<1><<<dim3(gemm_blocks(MBL, D_)), 256, 0, stream>>>(
            buf0, D_, 0, 0, wv, D_, 0, 0, buf3, D_, 0, 0, MBL, D_, D_, bv, 0);
        // sigma linear: [6464, 8]
        gemm_bf16_wmma<1><<<dim3(gemm_blocks(MBL, H_)), 256, 0, stream>>>(
            buf0, D_, 0, 0, wsig, D_, 0, 0, siglin, H_, 0, 0, MBL, H_, D_, bsig, 0);

        // scores[b,h] = q_head @ k_head^T  (M=N=101, K=64), batched over (b,h)
        gemm_bf16_wmma<1><<<dim3(gemm_blocks(L_, L_), B_, H_), 256, 0, stream>>>(
            buf1, D_, (long long)L_ * D_, E_,
            buf2, D_, (long long)L_ * D_, E_,
            series_l, L_, (long long)H_ * L_ * L_, (long long)L_ * L_,
            L_, L_, E_, nullptr, 0);
        softmax_kernel<<<B_ * H_ * L_, 128, 0, stream>>>(series_l);
        prior_sigma_kernel<<<B_ * H_ * L_, 128, 0, stream>>>(siglin, prior_l, sigma_l);

        // attnV[b,h] = series @ v_head (M=101, N=64, K=101) -> buf1 as [B,L,512]
        gemm_bf16_wmma<0><<<dim3(gemm_blocks(L_, E_), B_, H_), 256, 0, stream>>>(
            series_l, L_, (long long)H_ * L_ * L_, (long long)L_ * L_,
            buf3, D_, (long long)L_ * D_, E_,
            buf1, D_, (long long)L_ * D_, E_,
            L_, E_, L_, nullptr, 0);

        // new_x = attnV @ wo^T + bo -> buf2
        gemm_bf16_wmma<1><<<dim3(gemm_blocks(MBL, D_)), 256, 0, stream>>>(
            buf1, D_, 0, 0, wo, D_, 0, 0, buf2, D_, 0, 0, MBL, D_, D_, bo, 0);
        // xr = LN(e + new_x) -> buf3
        resid_ln_kernel<<<MBL, 256, 0, stream>>>(buf0, buf2, l1s, l1b, buf3);
        // y = gelu(xr @ w1^T + b1) -> buf1
        gemm_bf16_wmma<1><<<dim3(gemm_blocks(MBL, D_)), 256, 0, stream>>>(
            buf3, D_, 0, 0, w1, D_, 0, 0, buf1, D_, 0, 0, MBL, D_, D_, b1, 1);
        // y2 = y @ w2^T + b2 -> buf2
        gemm_bf16_wmma<1><<<dim3(gemm_blocks(MBL, D_)), 256, 0, stream>>>(
            buf1, D_, 0, 0, w2, D_, 0, 0, buf2, D_, 0, 0, MBL, D_, D_, b2, 0);
        // e = LN(xr + y2) -> buf0
        resid_ln_kernel<<<MBL, 256, 0, stream>>>(buf3, buf2, l2s, l2b, buf0);
    }

    head_kernel<<<B_, 256, 0, stream>>>(buf0, fin_s, fin_b, hln_s, hln_b,
                                        hw1, hb1, hw2, hb2, out);
}
